// MSCNL_19224273617286
// MI455X (gfx1250) — matrix operand order repaired and checked by reading, not compiled, not run on hardware
//
#include <hip/hip_runtime.h>
#include <math.h>

// ---------- WMMA fragment types / helpers (CDNA5 gfx1250, wave32) ----------
typedef __attribute__((ext_vector_type(8)))  __bf16 v8bf;
typedef __attribute__((ext_vector_type(16))) __bf16 v16bf;
typedef __attribute__((ext_vector_type(8)))  float  v8f;

#define DEVI static __device__ __forceinline__

DEVI unsigned short f2bf(float f) {            // fp32 -> bf16 (RNE)
    unsigned int u = __float_as_uint(f);
    u += 0x7FFFu + ((u >> 16) & 1u);
    return (unsigned short)(u >> 16);
}
DEVI v16bf cat16(v8bf lo, v8bf hi) {
    return __builtin_shufflevector(lo, hi, 0,1,2,3,4,5,6,7,8,9,10,11,12,13,14,15);
}
DEVI v8bf ldb8(const unsigned short* p) { return *(const v8bf*)p; }
DEVI v8bf zb8() {
    v8bf z;
#pragma unroll
    for (int i = 0; i < 8; ++i) z[i] = (__bf16)0.0f;
    return z;
}
DEVI v8f zf8() {
    v8f z;
#pragma unroll
    for (int i = 0; i < 8; ++i) z[i] = 0.0f;
    return z;
}
DEVI v8f wmma_bf(v16bf a, v16bf b, v8f c) {
    return __builtin_amdgcn_wmma_f32_16x16x32_bf16(false, a, false, b, (short)0, c, false, false);
}
DEVI void st8f(float* p, v8f v) {              // 8 contiguous f32 -> two b128 stores
    float4 a, b;
    a.x = v[0]; a.y = v[1]; a.z = v[2]; a.w = v[3];
    b.x = v[4]; b.y = v[5]; b.z = v[6]; b.w = v[7];
    *(float4*)p = a;
    *(float4*)(p + 4) = b;
}
DEVI void st8bf(unsigned short* p, v8f v) {    // 8 f32 -> 8 bf16 packed, one b128 store
    uint4 u;
    u.x = (unsigned int)f2bf(v[0]) | ((unsigned int)f2bf(v[1]) << 16);
    u.y = (unsigned int)f2bf(v[2]) | ((unsigned int)f2bf(v[3]) << 16);
    u.z = (unsigned int)f2bf(v[4]) | ((unsigned int)f2bf(v[5]) << 16);
    u.w = (unsigned int)f2bf(v[6]) | ((unsigned int)f2bf(v[7]) << 16);
    *(uint4*)p = u;
}

// Problem constants
#define NB   8
#define NPIX 16384      // 128*128

// ---------- prep kernels ----------
// q_w [128][64][3][3] f32 -> A [128][576] bf16, k = tap*64 + ic
__global__ __launch_bounds__(256) void pack_qw(const float* __restrict__ W,
                                               unsigned short* __restrict__ A) {
    int idx = blockIdx.x * 256 + threadIdx.x;
    if (idx >= 128 * 576) return;
    int k = idx % 576, m = idx / 576;
    int tap = k / 64, ic = k % 64;
    A[idx] = f2bf(W[((size_t)m * 64 + ic) * 9 + tap]);
}

// NCHW f32 [B][64][N] -> NHWC bf16 [B][N][64]
__global__ __launch_bounds__(256) void to_nhwc_bf16(const float* __restrict__ X,
                                                    unsigned short* __restrict__ Y) {
    size_t idx = (size_t)blockIdx.x * 256 + threadIdx.x; // over B*N*64
    int c   = (int)(idx & 63);
    int pix = (int)((idx >> 6) & (NPIX - 1));
    int b   = (int)(idx >> 20);
    Y[idx] = f2bf(X[((size_t)b * 64 + c) * NPIX + pix]);
}

// ---------- WMMA 3x3 conv (implicit GEMM, double-buffered LDS weights, pipelined frags) ----------
__global__ __launch_bounds__(256) void conv3x3_wmma(const unsigned short* __restrict__ Aw,  // [128][576]
                                                    const unsigned short* __restrict__ Xin, // [B][N][64]
                                                    unsigned short* __restrict__ Xout) {    // [B][N][128]
    const int H = 128, W = 128;
    const int APAD = 72;                          // padded LDS row stride (conflict-free)
    __shared__ unsigned short Alds[2][128 * APAD];  // 2 x 18.4 KB

    int blk  = blockIdx.x;       // b*128 + row
    int b    = blk >> 7;
    int row  = blk & 127;
    int tid  = threadIdx.x;
    int lane = tid & 31;
    int wave = tid >> 5;
    int nlo  = lane & 15;
    int hi   = lane >> 4;
    int x    = wave * 16 + nlo;

    v8f acc[8];
#pragma unroll
    for (int m = 0; m < 8; ++m) acc[m] = zf8();

    const unsigned short* xbase = Xin + (size_t)b * NPIX * 64;
    if (row + 1 < H) __builtin_prefetch(xbase + ((size_t)((row + 1) * W + x)) * 64, 0, 1);

    int lrow  = tid >> 1;        // cooperative-load row
    int lhalf = tid & 1;
    auto stage = [&](int tap, int buf) {
        const unsigned short* src = Aw + (size_t)lrow * 576 + tap * 64 + lhalf * 32;
        unsigned short* dst = &Alds[buf][lrow * APAD + lhalf * 32];
#pragma unroll
        for (int u = 0; u < 4; ++u) *(v8bf*)(dst + 8 * u) = ldb8(src + 8 * u);
    };

    stage(0, 0);
    __syncthreads();

#pragma unroll 1
    for (int tap = 0; tap < 9; ++tap) {
        int cb = tap & 1;
        int dy = tap / 3 - 1, dx = tap % 3 - 1;
        int sy = row + dy, sx = x + dx;
        bool inb = (sy >= 0 && sy < H && sx >= 0 && sx < W);

        // global B-fragment loads for both k-steps
        v16bf bfrag[2];
#pragma unroll
        for (int ks = 0; ks < 2; ++ks) {
            v8bf b0 = zb8(), b1 = zb8();
            if (inb) {
                const unsigned short* bp = xbase + ((size_t)(sy * W + sx)) * 64 + ks * 32 + 8 * hi;
                b0 = ldb8(bp);
                b1 = ldb8(bp + 16);
            }
            bfrag[ks] = cat16(b0, b1);
        }

        // stage next tap's weights into the other buffer (overlaps with WMMA below)
        if (tap < 8) stage(tap + 1, cb ^ 1);

        auto ldA = [&](int mt, int ks) -> v16bf {
            const unsigned short* ap = &Alds[cb][(mt * 16 + nlo) * APAD + ks * 32];
            return cat16(*(const v8bf*)(ap + 8 * hi), *(const v8bf*)(ap + 16 + 8 * hi));
        };
        // software-pipelined: issue ds_load for step i+1 before WMMA of step i
        v16bf cur = ldA(0, 0);
#pragma unroll
        for (int st = 0; st < 16; ++st) {
            int ks = st >> 3, mt = st & 7;
            v16bf nxt = cur;
            if (st < 15) nxt = ldA((st + 1) & 7, (st + 1) >> 3);
            acc[mt] = wmma_bf(cur, bfrag[ks], acc[mt]);
            cur = nxt;
        }
        __syncthreads();   // everyone done reading Alds[cb]; staged buffer visible next iter
    }
    unsigned short* op = Xout + ((size_t)b * NPIX + row * W + x) * 128;
#pragma unroll
    for (int mt = 0; mt < 8; ++mt) st8bf(op + mt * 16 + 8 * hi, acc[mt]);
}

// ---------- pyramid pooling (pool commutes with 1x1 conv; pad1 -> zero border) ----------
__global__ __launch_bounds__(128) void pap_pool(const float* __restrict__ F,  // [B][C][HH][HH]
                                                float* __restrict__ P,        // [B][C][110]
                                                int HHd) {
    int blk = blockIdx.x;   // b*C + c
    int t = threadIdx.x;
    if (t >= 110) return;
    int s, j;
    if (t == 0)      { s = 1; j = 0; }
    else if (t < 10) { s = 3; j = t - 1; }
    else if (t < 46) { s = 6; j = t - 10; }
    else             { s = 8; j = t - 46; }
    int jy = j / s, jx = j % s;
    int L = HHd + 2;
    int ay = (jy * L) / s, by = ((jy + 1) * L + s - 1) / s;
    int ax = (jx * L) / s, bx = ((jx + 1) * L + s - 1) / s;
    float sum = 0.f;
    for (int py = ay; py < by; ++py) {
        if (py < 1 || py > HHd) continue;
        for (int px = ax; px < bx; ++px) {
            if (px < 1 || px > HHd) continue;
            sum += F[((size_t)blk * HHd + (py - 1)) * HHd + (px - 1)];
        }
    }
    P[(size_t)blk * 110 + t] = sum / (float)((by - ay) * (bx - ax));
}

// xk/xv = W[128,C] @ pooled[C,110]; k -> transposed bf16 [B][112][128], v -> f32 [B][128][110]
__global__ __launch_bounds__(256) void small_mm_kv(const float* __restrict__ W,
                                                   const float* __restrict__ Pool, int C, int isK,
                                                   unsigned short* __restrict__ XkT,
                                                   float* __restrict__ Xv) {
    int idx = blockIdx.x * 256 + threadIdx.x;
    if (idx >= NB * 128 * 112) return;
    int p = idx % 112;
    int m = (idx / 112) % 128;
    int b = idx / (112 * 128);
    float s = 0.f;
    if (p < 110) {
        const float* pb = Pool + (size_t)b * C * 110 + p;
        const float* wr = W + (size_t)m * C;
        for (int c = 0; c < C; ++c) s += wr[c] * pb[(size_t)c * 110];
    }
    if (isK) XkT[((size_t)b * 112 + p) * 128 + m] = f2bf(p < 110 ? s : 0.f);
    else if (p < 110) Xv[((size_t)b * 128 + m) * 110 + p] = s;
}

// Wv[i][m][c] = sum_j fuse[m][64i+j] * mask[j][64+c]
__global__ __launch_bounds__(256) void fold_wv(const float* __restrict__ fuse,
                                               const float* __restrict__ mask,
                                               float* __restrict__ Wv) {
    int idx = blockIdx.x * 256 + threadIdx.x;
    if (idx >= 3 * 64 * 128) return;
    int c = idx % 128;
    int m = (idx / 128) % 64;
    int i = idx / (128 * 64);
    float s = 0.f;
    for (int j = 0; j < 64; ++j)
        s += fuse[(size_t)m * 192 + 64 * i + j] * mask[(size_t)j * 192 + 64 + c];
    Wv[idx] = s;
}

// Wg[m][c] = sum_j (sum_i fuse[m][64i+j]) * mask[j][c]  -> bf16 A-matrix
__global__ __launch_bounds__(256) void fold_wg(const float* __restrict__ fuse,
                                               const float* __restrict__ mask,
                                               unsigned short* __restrict__ Wg) {
    int idx = blockIdx.x * 256 + threadIdx.x;
    if (idx >= 64 * 64) return;
    int c = idx % 64, m = idx / 64;
    float s = 0.f;
    for (int j = 0; j < 64; ++j) {
        float fs = fuse[m * 192 + j] + fuse[m * 192 + 64 + j] + fuse[m * 192 + 128 + j];
        s += fs * mask[j * 192 + c];
    }
    Wg[idx] = f2bf(s);
}

// U[b][m][p] = sum_c Wv[m][c] * Xv[b][c][p]
__global__ __launch_bounds__(256) void fold_u(const float* __restrict__ Wv,
                                              const float* __restrict__ Xv,
                                              float* __restrict__ U) {
    int idx = blockIdx.x * 256 + threadIdx.x;
    if (idx >= NB * 64 * 112) return;
    int p = idx % 112;
    int m = (idx / 112) % 64;
    int b = idx / (112 * 64);
    float s = 0.f;
    if (p < 110)
        for (int c = 0; c < 128; ++c)
            s += Wv[(size_t)m * 128 + c] * Xv[((size_t)b * 128 + c) * 110 + p];
    U[idx] = s;
}

// ---------- WMMA: Acc[b][n][64] (NHWC) = Wg[64x64] @ X_nhwc ----------
__global__ __launch_bounds__(256) void wg_gemm_wmma(const unsigned short* __restrict__ Wg,  // [64][64]
                                                    const unsigned short* __restrict__ Xn,  // [B][N][64]
                                                    float* __restrict__ Acc) {              // [B][N][64]
    const int APAD = 72;
    __shared__ unsigned short Alds[64 * APAD];    // 9.2 KB
    int blk  = blockIdx.x;          // b*128 + nblk
    int b    = blk >> 7;
    int nblk = blk & 127;
    int tid  = threadIdx.x;
    int lane = tid & 31;
    int wave = tid >> 5;
    int nlo = lane & 15, hi = lane >> 4;
    int n = nblk * 128 + wave * 16 + nlo;
    const unsigned short* bp = Xn + ((size_t)b * NPIX + n) * 64;

    v16bf bfrag[2];
#pragma unroll
    for (int ks = 0; ks < 2; ++ks) {
        int kofs = ks * 32 + 8 * hi;
        bfrag[ks] = cat16(ldb8(bp + kofs), ldb8(bp + kofs + 16));
    }
    {
        int lrow = tid >> 2, lq = tid & 3;
        const unsigned short* src = Wg + (size_t)lrow * 64 + lq * 16;
        unsigned short* dst = &Alds[lrow * APAD + lq * 16];
        *(v8bf*)dst = ldb8(src);
        *(v8bf*)(dst + 8) = ldb8(src + 8);
    }
    __syncthreads();

    auto ldA = [&](int mt, int ks) -> v16bf {
        const unsigned short* ap = &Alds[(mt * 16 + nlo) * APAD + ks * 32];
        return cat16(*(const v8bf*)(ap + 8 * hi), *(const v8bf*)(ap + 16 + 8 * hi));
    };
    v8f acc[4];
#pragma unroll
    for (int m = 0; m < 4; ++m) acc[m] = zf8();
    v16bf cur = ldA(0, 0);
#pragma unroll
    for (int st = 0; st < 8; ++st) {
        int ks = st >> 2, mt = st & 3;
        v16bf nxt = cur;
        if (st < 7) nxt = ldA((st + 1) & 3, (st + 1) >> 2);
        acc[mt] = wmma_bf(cur, bfrag[ks], acc[mt]);
        cur = nxt;
    }
    float* op = Acc + ((size_t)b * NPIX + n) * 64;
#pragma unroll
    for (int mt = 0; mt < 4; ++mt) st8f(op + mt * 16 + 8 * hi, acc[mt]);
}

// ---------- WMMA: S[b][n][112] = xkT[b][112][128] @ xq[b][n][128]^T (A in LDS) ----------
__global__ __launch_bounds__(256) void kq_gemm_wmma(const unsigned short* __restrict__ Akt, // [B][112][128]
                                                    const unsigned short* __restrict__ Xq,  // [B][N][128]
                                                    float* __restrict__ S) {                // [B][N][112]
    const int APAD = 136;
    __shared__ unsigned short Alds[112 * APAD];   // 30.5 KB
    int blk  = blockIdx.x;
    int b    = blk >> 7;
    int nblk = blk & 127;
    int tid  = threadIdx.x;
    int lane = tid & 31;
    int wave = tid >> 5;
    int nlo = lane & 15, hi = lane >> 4;
    int n = nblk * 128 + wave * 16 + nlo;
    const unsigned short* abase = Akt + (size_t)b * 112 * 128;
    const unsigned short* bp = Xq + ((size_t)b * NPIX + n) * 128;

    // B fragments for all 4 k-steps up front
    v16bf bfrag[4];
#pragma unroll
    for (int ks = 0; ks < 4; ++ks) {
        int kofs = ks * 32 + 8 * hi;
        bfrag[ks] = cat16(ldb8(bp + kofs), ldb8(bp + kofs + 16));
    }
    // stage whole A tile (112x128) in LDS
    if (tid < 224) {
        int lrow = tid >> 1, lhalf = tid & 1;
        const unsigned short* src = abase + (size_t)lrow * 128 + lhalf * 64;
        unsigned short* dst = &Alds[lrow * APAD + lhalf * 64];
#pragma unroll
        for (int u = 0; u < 8; ++u) *(v8bf*)(dst + 8 * u) = ldb8(src + 8 * u);
    }
    __syncthreads();

    auto ldA = [&](int mt, int ks) -> v16bf {
        const unsigned short* ap = &Alds[(mt * 16 + nlo) * APAD + ks * 32];
        return cat16(*(const v8bf*)(ap + 8 * hi), *(const v8bf*)(ap + 16 + 8 * hi));
    };
    v8f acc[7];
#pragma unroll
    for (int m = 0; m < 7; ++m) acc[m] = zf8();
    v16bf cur = ldA(0, 0);
#pragma unroll
    for (int st = 0; st < 28; ++st) {
        int ks = st / 7, mt = st % 7;
        v16bf nxt = cur;
        if (st < 27) nxt = ldA((st + 1) % 7, (st + 1) / 7);
        acc[mt] = wmma_bf(cur, bfrag[ks], acc[mt]);
        cur = nxt;
    }
    float* op = S + ((size_t)b * NPIX + n) * 112;
#pragma unroll
    for (int mt = 0; mt < 7; ++mt) st8f(op + mt * 16 + 8 * hi, acc[mt]);
}

// ---------- fused clip+softmax over 110 + U@softmax accumulate (NHWC acc) ----------
__global__ __launch_bounds__(256) void softmax_u_acc(const float* __restrict__ S,  // [B][N][112]
                                                     const float* __restrict__ U,  // [B][64][112]
                                                     float* __restrict__ Acc) {    // [B][N][64]
    __shared__ float uls[64 * 112];
    int tid = threadIdx.x;
    size_t q = (size_t)blockIdx.x * 256 + tid;   // 131072 pixels total (block within one batch)
    int b = (int)(q >> 14);
    const float* ub = U + (size_t)b * 64 * 112;
    for (int i = tid; i < 64 * 112; i += 256) uls[i] = ub[i];
    __syncthreads();

    const float* sp = S + q * 112;
    float esum = 0.f;
    for (int p = 0; p < 110; ++p) {
        float v = fminf(64.f, fmaxf(-64.f, sp[p]));
        esum += __expf(v);
    }
    float inv = 1.f / esum;
    float accm[64];
#pragma unroll
    for (int m = 0; m < 64; ++m) accm[m] = 0.f;
    for (int p = 0; p < 110; ++p) {
        float v = fminf(64.f, fmaxf(-64.f, sp[p]));
        float e = __expf(v);
#pragma unroll
        for (int m = 0; m < 64; ++m) accm[m] += uls[m * 112 + p] * e;
    }
    float* ap = Acc + q * 64;
#pragma unroll
    for (int m0 = 0; m0 < 64; m0 += 4) {
        float4 v = *(float4*)(ap + m0);
        v.x += accm[m0 + 0] * inv;
        v.y += accm[m0 + 1] * inv;
        v.z += accm[m0 + 2] * inv;
        v.w += accm[m0 + 3] * inv;
        *(float4*)(ap + m0) = v;
    }
}

// x = tanh(acc) + input ; acc NHWC, input NCHW, out NHWC f32
__global__ __launch_bounds__(256) void tanh_res(const float* __restrict__ Acc,  // [B][N][64]
                                                const float* __restrict__ Xin,  // NCHW
                                                float* __restrict__ Xout) {     // [B][N][64]
    size_t idx = (size_t)blockIdx.x * 256 + threadIdx.x;  // NHWC flat
    int c   = (int)(idx & 63);
    int pix = (int)((idx >> 6) & (NPIX - 1));
    int b   = (int)(idx >> 20);
    Xout[idx] = tanhf(Acc[idx]) + Xin[((size_t)b * 64 + c) * NPIX + pix];
}

// ---------- BNL: s0,s1,s2 per (b,group) from p=p_w@xc, g=g_w@xc ----------
__global__ __launch_bounds__(256) void bnl_pg_reduce(const float* __restrict__ Xc,  // [B][N][64]
                                                     const float* __restrict__ PW,
                                                     const float* __restrict__ GW,
                                                     float* __restrict__ Part) {    // [B][8][16][3]
    const int CHUNK = 16, PIXPER = NPIX / CHUNK;
    int blk = blockIdx.x;
    int ck = blk % CHUNK;
    int g  = (blk / CHUNK) % 8;
    int b  = blk / (CHUNK * 8);
    __shared__ float pwl[512], gwl[512];
    __shared__ float r0[256], r1[256], r2[256];
    int tid = threadIdx.x;
    for (int i = tid; i < 512; i += 256) {
        pwl[i] = PW[(g * 8 + i / 64) * 64 + (i % 64)];
        gwl[i] = GW[(g * 8 + i / 64) * 64 + (i % 64)];
    }
    __syncthreads();
    float s0 = 0.f, s1 = 0.f, s2 = 0.f;
    for (int pp = tid; pp < PIXPER; pp += 256) {
        int pix = ck * PIXPER + pp;
        const float* xp = Xc + ((size_t)b * NPIX + pix) * 64;
#pragma unroll
        for (int ch = 0; ch < 8; ++ch) {
            float pv = 0.f, gv = 0.f;
            for (int c = 0; c < 64; ++c) {
                float xv = xp[c];
                pv += pwl[ch * 64 + c] * xv;
                gv += gwl[ch * 64 + c] * xv;
            }
            s0 += gv; s1 += pv * gv; s2 += pv * pv * gv;
        }
    }
    r0[tid] = s0; r1[tid] = s1; r2[tid] = s2;
    __syncthreads();
    for (int o = 128; o > 0; o >>= 1) {
        if (tid < o) { r0[tid] += r0[tid + o]; r1[tid] += r1[tid + o]; r2[tid] += r2[tid + o]; }
        __syncthreads();
    }
    if (tid == 0) {
        float* op = Part + ((size_t)(b * 8 + g) * 16 + ck) * 3;
        op[0] = r0[0]; op[1] = r1[0]; op[2] = r2[0];
    }
}

__global__ void bnl_coef(const float* __restrict__ Part, float* __restrict__ Coef,
                         float c0, float c1, float c2) {
    int idx = blockIdx.x * blockDim.x + threadIdx.x;
    if (idx >= 64) return;
    float s0 = 0.f, s1 = 0.f, s2 = 0.f;
    for (int k = 0; k < 16; ++k) {
        const float* p = Part + ((size_t)idx * 16 + k) * 3;
        s0 += p[0]; s1 += p[1]; s2 += p[2];
    }
    Coef[idx * 3 + 0] = c0 * s0;
    Coef[idx * 3 + 1] = c1 * s1;
    Coef[idx * 3 + 2] = c2 * s2;
}

// y = zconv(c0 + c1*t + c2*t^2), t = t_w@x_gray ; accumulate group sum/sumsq
__global__ __launch_bounds__(256) void bnl_y(const float* __restrict__ Xg,    // [B][N][64]
                                             const float* __restrict__ TW,
                                             const float* __restrict__ ZW,    // [8][8][8]
                                             const float* __restrict__ Coef,  // [B][8][3]
                                             float* __restrict__ Y,           // [B][N][64]
                                             float* __restrict__ Part2) {     // [B][8][16][2]
    const int CHUNK = 16, PIXPER = NPIX / CHUNK;
    int blk = blockIdx.x;
    int ck = blk % CHUNK;
    int g  = (blk / CHUNK) % 8;
    int b  = blk / (CHUNK * 8);
    __shared__ float twl[512], zwl[64];
    __shared__ float r0[256], r1[256];
    int tid = threadIdx.x;
    for (int i = tid; i < 512; i += 256) twl[i] = TW[(g * 8 + i / 64) * 64 + (i % 64)];
    if (tid < 64) zwl[tid] = ZW[g * 64 + tid];
    __syncthreads();
    float c0 = Coef[(b * 8 + g) * 3 + 0];
    float c1 = Coef[(b * 8 + g) * 3 + 1];
    float c2 = Coef[(b * 8 + g) * 3 + 2];
    float sy = 0.f, sy2 = 0.f;
    for (int pp = tid; pp < PIXPER; pp += 256) {
        int pix = ck * PIXPER + pp;
        const float* xp = Xg + ((size_t)b * NPIX + pix) * 64;
        float xv[8];
#pragma unroll
        for (int ch = 0; ch < 8; ++ch) {
            float t = 0.f;
            for (int c = 0; c < 64; ++c) t += twl[ch * 64 + c] * xp[c];
            xv[ch] = c0 + c1 * t + c2 * t * t;
        }
        float* yp = Y + ((size_t)b * NPIX + pix) * 64 + g * 8;
#pragma unroll
        for (int o = 0; o < 8; ++o) {
            float yv = 0.f;
#pragma unroll
            for (int i = 0; i < 8; ++i) yv += zwl[o * 8 + i] * xv[i];
            yp[o] = yv;
            sy += yv; sy2 += yv * yv;
        }
    }
    r0[tid] = sy; r1[tid] = sy2;
    __syncthreads();
    for (int o = 128; o > 0; o >>= 1) {
        if (tid < o) { r0[tid] += r0[tid + o]; r1[tid] += r1[tid + o]; }
        __syncthreads();
    }
    if (tid == 0) {
        float* op = Part2 + ((size_t)(b * 8 + g) * 16 + ck) * 2;
        op[0] = r0[0]; op[1] = r1[0];
    }
}

__global__ void bnl_stats(const float* __restrict__ Part2, float* __restrict__ MuR) {
    int idx = blockIdx.x * blockDim.x + threadIdx.x;
    if (idx >= 64) return;
    float s = 0.f, s2 = 0.f;
    for (int k = 0; k < 16; ++k) {
        const float* p = Part2 + ((size_t)idx * 16 + k) * 2;
        s += p[0]; s2 += p[1];
    }
    const float inv = 1.0f / 131072.0f;
    float mu = s * inv;
    float var = s2 * inv - mu * mu;
    MuR[idx * 2 + 0] = mu;
    MuR[idx * 2 + 1] = rsqrtf(var + 1e-5f);
}

__global__ __launch_bounds__(256) void bnl_final(const float* __restrict__ Y,
                                                 const float* __restrict__ MuR,
                                                 const float* __restrict__ GNW,
                                                 const float* __restrict__ GNB,
                                                 const float* __restrict__ Xg,   // [B][N][64]
                                                 const float* __restrict__ InG,  // NCHW
                                                 float* __restrict__ Out) {
    size_t idx = (size_t)blockIdx.x * 256 + threadIdx.x;  // NCHW flat
    int pix = (int)(idx & (NPIX - 1));
    int ch  = (int)((idx >> 14) & 63);
    int b   = (int)(idx >> 20);
    int g = ch >> 3;
    float y  = Y[((size_t)b * NPIX + pix) * 64 + ch];
    float mu = MuR[(size_t)(b * 8 + g) * 2 + 0];
    float rs = MuR[(size_t)(b * 8 + g) * 2 + 1];
    float xn = (y - mu) * rs * GNW[ch] + GNB[ch];
    Out[idx] = xn + Xg[((size_t)b * NPIX + pix) * 64 + ch] + InG[idx];
}

// --------------------------------------------------------------------------
extern "C" void kernel_launch(void* const* d_in, const int* in_sizes, int n_in,
                              void* d_out, int out_size, void* d_ws, size_t ws_size,
                              hipStream_t stream) {
    (void)in_sizes; (void)n_in; (void)out_size; (void)ws_size;
    const float* in_g = (const float*)d_in[0];
    const float* in_c = (const float*)d_in[1];
    const float* fgp[3] = {(const float*)d_in[2], (const float*)d_in[8],  (const float*)d_in[14]};
    const float* fcp[3] = {(const float*)d_in[3], (const float*)d_in[9],  (const float*)d_in[15]};
    const float* k1w[3] = {(const float*)d_in[4], (const float*)d_in[10], (const float*)d_in[16]};
    const float* v1w[3] = {(const float*)d_in[5], (const float*)d_in[11], (const float*)d_in[17]};
    const float* k2w[3] = {(const float*)d_in[6], (const float*)d_in[12], (const float*)d_in[18]};
    const float* v2w[3] = {(const float*)d_in[7], (const float*)d_in[13], (const float*)d_in[19]};
    const float* q1w   = (const float*)d_in[20];
    const float* q2w   = (const float*)d_in[21];
    const float* mask1 = (const float*)d_in[22];
    const float* mask2 = (const float*)d_in[23];
    const float* fuse1 = (const float*)d_in[24];
    const float* fuse2 = (const float*)d_in[25];
    const float* tw = (const float*)d_in[26];
    const float* pw = (const float*)d_in[27];
    const float* gw = (const float*)d_in[28];
    const float* zw = (const float*)d_in[29];
    const float* gnw = (const float*)d_in[30];
    const float* gnb = (const float*)d_in[31];
    float* out = (float*)d_out;

    const int CS[3] = {64, 128, 256};
    const int HS[3] = {128, 64, 32};

    // workspace bump allocator
    char* base = (char*)d_ws;
    size_t off = 0;
    auto alloc = [&](size_t bytes) -> void* {
        void* p = base + off;
        off = (off + bytes + 255) & ~(size_t)255;
        return p;
    };
    unsigned short* xg_nh = (unsigned short*)alloc((size_t)NB * NPIX * 64 * 2);
    unsigned short* xc_nh = (unsigned short*)alloc((size_t)NB * NPIX * 64 * 2);
    unsigned short* Aq1 = (unsigned short*)alloc((size_t)128 * 576 * 2);
    unsigned short* Aq2 = (unsigned short*)alloc((size_t)128 * 576 * 2);
    unsigned short* xq1 = (unsigned short*)alloc((size_t)NB * NPIX * 128 * 2);
    unsigned short* xq2 = (unsigned short*)alloc((size_t)NB * NPIX * 128 * 2);
    float* pool_fc[3]; float* pool_fg[3];
    for (int i = 0; i < 3; ++i) {
        pool_fc[i] = (float*)alloc((size_t)NB * CS[i] * 110 * 4);
        pool_fg[i] = (float*)alloc((size_t)NB * CS[i] * 110 * 4);
    }
    unsigned short* XkT1[3]; unsigned short* XkT2[3];
    float* Xv1[3]; float* Xv2[3];
    for (int i = 0; i < 3; ++i) {
        XkT1[i] = (unsigned short*)alloc((size_t)NB * 112 * 128 * 2);
        XkT2[i] = (unsigned short*)alloc((size_t)NB * 112 * 128 * 2);
        Xv1[i] = (float*)alloc((size_t)NB * 128 * 110 * 4);
        Xv2[i] = (float*)alloc((size_t)NB * 128 * 110 * 4);
    }
    unsigned short* Wg1 = (unsigned short*)alloc(64 * 64 * 2);
    unsigned short* Wg2 = (unsigned short*)alloc(64 * 64 * 2);
    float* Wv1 = (float*)alloc((size_t)3 * 64 * 128 * 4);
    float* Wv2 = (float*)alloc((size_t)3 * 64 * 128 * 4);
    float* U1[3]; float* U2[3];
    for (int i = 0; i < 3; ++i) {
        U1[i] = (float*)alloc((size_t)NB * 64 * 112 * 4);
        U2[i] = (float*)alloc((size_t)NB * 64 * 112 * 4);
    }
    float* Sbuf  = (float*)alloc((size_t)NB * NPIX * 112 * 4);
    float* acc_g = (float*)alloc((size_t)NB * NPIX * 64 * 4);
    float* acc_c = (float*)alloc((size_t)NB * NPIX * 64 * 4);
    float* xg_f  = (float*)alloc((size_t)NB * NPIX * 64 * 4);
    float* xc_f  = (float*)alloc((size_t)NB * NPIX * 64 * 4);
    float* Ybuf  = (float*)alloc((size_t)NB * NPIX * 64 * 4);
    float* Part  = (float*)alloc((size_t)NB * 8 * 16 * 3 * 4);
    float* Coef  = (float*)alloc((size_t)NB * 8 * 3 * 4);
    float* Part2 = (float*)alloc((size_t)NB * 8 * 16 * 2 * 4);
    float* MuR   = (float*)alloc((size_t)NB * 8 * 2 * 4);

    // 1. pack conv weights, NHWC transposes
    pack_qw<<<288, 256, 0, stream>>>(q1w, Aq1);
    pack_qw<<<288, 256, 0, stream>>>(q2w, Aq2);
    to_nhwc_bf16<<<32768, 256, 0, stream>>>(in_g, xg_nh);
    to_nhwc_bf16<<<32768, 256, 0, stream>>>(in_c, xc_nh);

    // 2. WMMA 3x3 convs
    conv3x3_wmma<<<1024, 256, 0, stream>>>(Aq1, xg_nh, xq1);
    conv3x3_wmma<<<1024, 256, 0, stream>>>(Aq2, xc_nh, xq2);

    // 3. pooling + small k/v GEMMs + U folding
    for (int i = 0; i < 3; ++i) {
        pap_pool<<<NB * CS[i], 128, 0, stream>>>(fcp[i], pool_fc[i], HS[i]);
        pap_pool<<<NB * CS[i], 128, 0, stream>>>(fgp[i], pool_fg[i], HS[i]);
        small_mm_kv<<<448, 256, 0, stream>>>(k1w[i], pool_fc[i], CS[i], 1, XkT1[i], nullptr);
        small_mm_kv<<<448, 256, 0, stream>>>(v1w[i], pool_fc[i], CS[i], 0, nullptr, Xv1[i]);
        small_mm_kv<<<448, 256, 0, stream>>>(k2w[i], pool_fg[i], CS[i], 1, XkT2[i], nullptr);
        small_mm_kv<<<448, 256, 0, stream>>>(v2w[i], pool_fg[i], CS[i], 0, nullptr, Xv2[i]);
    }
    fold_wv<<<96, 256, 0, stream>>>(fuse1, mask1, Wv1);
    fold_wv<<<96, 256, 0, stream>>>(fuse2, mask2, Wv2);
    fold_wg<<<16, 256, 0, stream>>>(fuse1, mask1, Wg1);
    fold_wg<<<16, 256, 0, stream>>>(fuse2, mask2, Wg2);
    for (int i = 0; i < 3; ++i) {
        fold_u<<<224, 256, 0, stream>>>(Wv1 + (size_t)i * 64 * 128, Xv1[i], U1[i]);
        fold_u<<<224, 256, 0, stream>>>(Wv2 + (size_t)i * 64 * 128, Xv2[i], U2[i]);
    }

    // 4. init acc = Wg @ input (WMMA), then per-scale score GEMM + fused softmax+U
    wg_gemm_wmma<<<1024, 256, 0, stream>>>(Wg1, xg_nh, acc_g);
    wg_gemm_wmma<<<1024, 256, 0, stream>>>(Wg2, xc_nh, acc_c);
    for (int i = 0; i < 3; ++i) {
        kq_gemm_wmma<<<1024, 256, 0, stream>>>(XkT1[i], xq1, Sbuf);
        softmax_u_acc<<<512, 256, 0, stream>>>(Sbuf, U1[i], acc_g);
        kq_gemm_wmma<<<1024, 256, 0, stream>>>(XkT2[i], xq2, Sbuf);
        softmax_u_acc<<<512, 256, 0, stream>>>(Sbuf, U2[i], acc_c);
    }

    // 5. x_gray / x_color = tanh(acc) + input (NHWC f32)
    tanh_res<<<32768, 256, 0, stream>>>(acc_g, in_g, xg_f);
    tanh_res<<<32768, 256, 0, stream>>>(acc_c, in_c, xc_f);

    // 6. BNL: Taylor-kernel attention reduces to 3 scalars per (b,group)
    const double g2 = 2.0 * 1e-4;
    const double eg = exp(-g2);
    float c0 = (float)eg, c1 = (float)(g2 * eg), c2 = (float)(0.5 * g2 * g2 * eg);
    bnl_pg_reduce<<<1024, 256, 0, stream>>>(xc_f, pw, gw, Part);
    bnl_coef<<<1, 64, 0, stream>>>(Part, Coef, c0, c1, c2);
    bnl_y<<<1024, 256, 0, stream>>>(xg_f, tw, zw, Coef, Ybuf, Part2);
    bnl_stats<<<1, 64, 0, stream>>>(Part2, MuR);
    bnl_final<<<32768, 256, 0, stream>>>(Ybuf, MuR, gnw, gnb, xg_f, in_g, out);
}
